// PTA_EGNN_17772574671127
// MI455X (gfx1250) — matrix operand order repaired and checked by reading, not compiled, not run on hardware
//
#include <hip/hip_runtime.h>

typedef __attribute__((ext_vector_type(16))) _Float16 v16h;
typedef __attribute__((ext_vector_type(8)))  _Float16 h8;
typedef __attribute__((ext_vector_type(8)))  float    v8f;

#define HIDN 128
#define DEGN 12

__device__ __forceinline__ int   ring_off(int i) { return (i < 6) ? (i - 6) : (i - 5); }
__device__ __forceinline__ int   ring_idx(int d) { return (d < 0) ? (d + 6) : (d + 5); }
__device__ __forceinline__ float siluf(float x)  { return x / (1.f + __expf(-x)); }

// ---------------------------------------------------------------------------
// Tensor Data Mover: 2-D tile load Global(f16) -> LDS, D# per ISA 8.3/8.4.
// Toolchain probe: 6-arg clang-23 form (g0 u32x4, g1 i32x8, g2 i32x4,
// g3 i32x4, extra i32x8, cpol i32).
// ---------------------------------------------------------------------------
#if defined(__gfx1250__) && __has_builtin(__builtin_amdgcn_tensor_load_to_lds)
#define HAVE_TDM 1
typedef unsigned int u32x4 __attribute__((ext_vector_type(4)));
typedef int          i32x4 __attribute__((ext_vector_type(4)));
typedef int          i32x8 __attribute__((ext_vector_type(8)));

__device__ __forceinline__ void tdm_load_tile_f16(
    unsigned lds_off, const void* gptr,
    unsigned tile_x_elems, unsigned tile_y,
    unsigned tensor_x_elems, unsigned tensor_y,
    unsigned row_stride_elems)
{
  unsigned long long ga = (unsigned long long)(size_t)gptr;
  u32x4 g0;
  g0[0] = 1u;                                   // count=1 (valid user D#)
  g0[1] = lds_off;                              // lds_addr (bytes)
  g0[2] = (unsigned)(ga & 0xffffffffu);         // global_addr[31:0]
  g0[3] = (unsigned)(ga >> 32) | (2u << 30);    // global_addr[56:32] | type=2
  i32x8 g1;
  g1[0] = (int)(1u << 16);                      // data_size=1 -> 2-byte elems
  g1[1] = (int)((tensor_x_elems & 0xffffu) << 16);                 // dim0 lo
  g1[2] = (int)((tensor_x_elems >> 16) | ((tensor_y & 0xffffu) << 16));
  g1[3] = (int)((tensor_y >> 16) | (tile_x_elems << 16));          // tile_dim0
  g1[4] = (int)(tile_y & 0xffffu);              // tile_dim1 (tile_dim2=0)
  g1[5] = (int)row_stride_elems;                // tensor_dim0_stride lo32
  g1[6] = 0;                                    // stride hi / dim1_stride lo
  g1[7] = 0;
  i32x4 z4 = {0, 0, 0, 0};
  i32x8 z8 = {0, 0, 0, 0, 0, 0, 0, 0};
  __builtin_amdgcn_tensor_load_to_lds(g0, g1, z4, z4, z8, 0);
}
#endif

// ---------------------------------------------------------------------------
// Fused GEMM:  C[M,128] = act(A[M,Ka(f16)] @ W[K,128](f32) + bias)
// act: 0=none 1=silu 2=sigmoid. 256 threads = 8 waves; 32-row M tile;
// each wave owns one 16-col N tile and issues 2 WMMAs per 32-K chunk
// (reusing its B fragment). A tile staged by the TDM, W tile staged+
// transposed+f32->f16 by all threads.
// ---------------------------------------------------------------------------
__global__ __launch_bounds__(256) void k_gemm_n128(
    const _Float16* __restrict__ A, int lda, int Ka,
    const float* __restrict__ W, int K,
    const float* __restrict__ bias, int act,
    float* __restrict__ Cf, _Float16* __restrict__ Ch,
    int ldc, int coloff, int M)
{
  __shared__ _Float16 As[32][32];    // 2 KB
  __shared__ _Float16 Wt[128][32];   // 8 KB transposed weight tile

  const int tid  = threadIdx.x;
  const int wave = tid >> 5;
  const int lane = tid & 31;
  const int nloc = lane & 15;
  const int kh   = lane >> 4;
  const int m0   = blockIdx.x * 32;

  v8f c0 = {}, c1 = {};

  for (int k0 = 0; k0 < Ka; k0 += 32) {
    // ---- stage A tile 32x32 f16 ----
#if defined(HAVE_TDM)
    if (wave == 0) {
      tdm_load_tile_f16((unsigned)(size_t)&As[0][0],
                        A + (size_t)m0 * lda + k0,
                        32u, 32u,
                        (unsigned)(Ka - k0), (unsigned)(M - m0),
                        (unsigned)lda);
    }
#else
    {
      int r  = tid >> 4;          // 0..15
      int dc = tid & 15;          // u32 column
#pragma unroll
      for (int it = 0; it < 2; ++it) {
        int rr = r + it * 16;
        int gm = m0 + rr;
        unsigned v = 0u;
        if (gm < M) v = *(const unsigned*)(A + (size_t)gm * lda + k0 + dc * 2);
        *(unsigned*)(&As[rr][dc * 2]) = v;
      }
    }
#endif
    // ---- stage W tile transposed (guard hoisted: K%16==0, 16-run uniform) ----
    {
      int n  = tid & 127;
      int hs = tid >> 7;
      int kb = k0 + hs * 16;
      if (kb < K) {
#pragma unroll
        for (int t = 0; t < 16; ++t)
          Wt[n][hs * 16 + t] = (_Float16)W[(size_t)(kb + t) * HIDN + n];
      } else {
#pragma unroll
        for (int t = 0; t < 16; ++t)
          Wt[n][hs * 16 + t] = (_Float16)0.f;
      }
      if (k0 + 32 < K)
        __builtin_prefetch(W + (size_t)(k0 + 32) * HIDN + n, 0, 0);
    }
#if defined(HAVE_TDM)
    if (wave == 0) __builtin_amdgcn_s_wait_tensorcnt(0);
#endif
    __syncthreads();

    // ---- fragments per ISA 7.12.2, two WMMAs sharing B ----
    h8 bl = *(const h8*)&Wt[wave * 16 + nloc][kh * 8];
    h8 bh = *(const h8*)&Wt[wave * 16 + nloc][16 + kh * 8];
    h8 al0 = *(const h8*)&As[nloc][kh * 8];
    h8 ah0 = *(const h8*)&As[nloc][16 + kh * 8];
    h8 al1 = *(const h8*)&As[16 + nloc][kh * 8];
    h8 ah1 = *(const h8*)&As[16 + nloc][16 + kh * 8];
    v16h a0, a1, b;
#pragma unroll
    for (int i = 0; i < 8; ++i) {
      a0[i] = al0[i]; a0[8 + i] = ah0[i];
      a1[i] = al1[i]; a1[8 + i] = ah1[i];
      b[i]  = bl[i];  b[8 + i]  = bh[i];
    }
    c0 = __builtin_amdgcn_wmma_f32_16x16x32_f16(false, a0, false, b, (short)0, c0, false, false);
    c1 = __builtin_amdgcn_wmma_f32_16x16x32_f16(false, a1, false, b, (short)0, c1, false, false);
    __syncthreads();
  }

  // ---- epilogue: bias + activation + store (two 16-row tiles) ----
  const int col = coloff + wave * 16 + nloc;
  float bv = bias ? bias[wave * 16 + nloc] : 0.f;
#pragma unroll
  for (int r = 0; r < 8; ++r) {
#pragma unroll
    for (int it = 0; it < 2; ++it) {
      int gm = m0 + it * 16 + r + 8 * kh;
      if (gm >= M) continue;
      float x = (it == 0 ? c0[r] : c1[r]) + bv;
      if (act == 1)      x = siluf(x);
      else if (act == 2) x = 1.f / (1.f + __expf(-x));
      if (Cf) Cf[(size_t)gm * ldc + col] = x;
      if (Ch) Ch[(size_t)gm * ldc + col] = (_Float16)x;
    }
  }
}

// ---------------------------------------------------------------------------
// Per-edge prep: geometry, pos-MLP layer1 (K=2, VALU), chem_in gather (f16,
// padded to 544 cols). Per-segment copies -> uniform, coalesced.
// ---------------------------------------------------------------------------
__global__ __launch_bounds__(128) void k_prep(
    const float* __restrict__ h, const float* __restrict__ coord,
    const float* __restrict__ nvecs, const float* __restrict__ edge_attr,
    const float* __restrict__ node_attr, const int* __restrict__ edges,
    const float* __restrict__ pe_w1, const float* __restrict__ pe_b1,
    _Float16* __restrict__ chem_in, _Float16* __restrict__ posh1,
    float* __restrict__ cdiff, int E)
{
  int e = blockIdx.x, j = threadIdx.x;
  int r = edges[e], cc = edges[E + e];
  float dx = coord[r * 3 + 0] - coord[cc * 3 + 0];
  float dy = coord[r * 3 + 1] - coord[cc * 3 + 1];
  float dz = coord[r * 3 + 2] - coord[cc * 3 + 2];
  float radial = dx * dx + dy * dy + dz * dz;
  float inv = 1.f / (__builtin_sqrtf(radial) + 1e-8f);
  if (j < 3) cdiff[(size_t)e * 3 + j] = (j == 0 ? dx : j == 1 ? dy : dz) * inv;
  float np = nvecs[r * 3 + 0] * nvecs[cc * 3 + 0] + nvecs[r * 3 + 1] * nvecs[cc * 3 + 1] +
             nvecs[r * 3 + 2] * nvecs[cc * 3 + 2];
  float x = np * pe_w1[j] + radial * pe_w1[HIDN + j] + pe_b1[j];
  posh1[(size_t)e * HIDN + j] = (_Float16)siluf(x);

  _Float16* ci = chem_in + (size_t)e * 544;
  ci[j]       = (_Float16)h[(size_t)r * 128 + j];
  ci[128 + j] = (_Float16)h[(size_t)cc * 128 + j];
  ci[256 + j] = (_Float16)node_attr[(size_t)r * 128 + j];
  ci[384 + j] = (_Float16)node_attr[(size_t)cc * 128 + j];
  if (j < 32) ci[512 + j] = (j < 16) ? (_Float16)edge_attr[(size_t)e * 16 + j] : (_Float16)0.f;
}

// z = (silu_sh * pos) gated by sigmoid(z @ att_w) ; one block per edge
__global__ __launch_bounds__(128) void k_zgate(
    const float* __restrict__ tsh, const float* __restrict__ pos,
    const float* __restrict__ att_w, _Float16* __restrict__ z16)
{
  __shared__ float red[128];
  int e = blockIdx.x, j = threadIdx.x;
  float z0 = tsh[(size_t)e * HIDN + j] * pos[(size_t)e * HIDN + j];
  red[j] = z0 * att_w[j];
  __syncthreads();
  for (int s = 64; s > 0; s >>= 1) { if (j < s) red[j] += red[j + s]; __syncthreads(); }
  float gate = 1.f / (1.f + __expf(-red[0]));
  z16[(size_t)e * HIDN + j] = (_Float16)(z0 * gate);
}

// b[e,h] = z[e,:] . Wb[h,:]
__global__ __launch_bounds__(256) void k_bfeat(
    const _Float16* __restrict__ z16, const float* __restrict__ Wb,
    float* __restrict__ bfeat, int E)
{
  int idx = blockIdx.x * 256 + threadIdx.x;
  if (idx >= E * 4) return;
  int e = idx >> 2, hh = idx & 3;
  const _Float16* zp = z16 + (size_t)e * HIDN;
  float acc = 0.f;
#pragma unroll 4
  for (int f8 = 0; f8 < 16; ++f8) {
    h8 zv = *(const h8*)(zp + f8 * 8);
#pragma unroll
    for (int t = 0; t < 8; ++t) acc += (float)zv[t] * Wb[hh * HIDN + f8 * 8 + t];
  }
  bfeat[idx] = acc;
}

// Triangle attention phase A: per (edge, head), analytic ring-graph triangle
// enumeration -> softmax over <=11 neighbors, store normalized weights.
__global__ __launch_bounds__(256) void k_triA(
    const _Float16* __restrict__ q16, const _Float16* __restrict__ k16,
    const float* __restrict__ bfeat, float* __restrict__ pnorm, int E, int Nn)
{
  int idx = blockIdx.x * 256 + threadIdx.x;
  if (idx >= E * 4) return;
  int e = idx >> 2, hh = idx & 3;
  int i = e / DEGN, di = e % DEGN;
  int d = ring_off(di);
  int jn = i + d; if (jn < 0) jn += Nn; if (jn >= Nn) jn -= Nn;
  const _Float16* qp = q16 + (size_t)e * 512 + hh * HIDN;
  const float scale = 0.08838834764831845f; // 1/sqrt(128)
  float s[12], p[12];
  float smax = -1e30f;
#pragma unroll
  for (int ci = 0; ci < 12; ++ci) {
    int dd = ring_off(ci) - d;
    float sv = -1e30f;
    if (dd != 0 && dd >= -6 && dd <= 6) {
      int ik = i * DEGN + ci;
      int jk = jn * DEGN + ring_idx(dd);
      const _Float16* kp = k16 + (size_t)ik * 512 + hh * HIDN;
      float acc = 0.f;
      for (int f8 = 0; f8 < 16; ++f8) {
        h8 qa = *(const h8*)(qp + f8 * 8);
        h8 ka = *(const h8*)(kp + f8 * 8);
#pragma unroll
        for (int t = 0; t < 8; ++t) acc += (float)qa[t] * (float)ka[t];
      }
      sv = acc * scale + bfeat[jk * 4 + hh];
      if (sv > smax) smax = sv;
    }
    s[ci] = sv;
  }
  float den = 0.f;
#pragma unroll
  for (int ci = 0; ci < 12; ++ci) {
    float pv = (s[ci] > -1e29f) ? __expf(s[ci] - smax) : 0.f;
    p[ci] = pv; den += pv;
  }
  float rden = (den > 0.f) ? 1.f / den : 0.f;
#pragma unroll
  for (int ci = 0; ci < 12; ++ci) pnorm[(size_t)idx * 12 + ci] = p[ci] * rden;
}

// Phase B: per (edge, feature) weighted v-gather, gate, pack m_in [E,512] f16
__global__ __launch_bounds__(256) void k_triB(
    const _Float16* __restrict__ v16x, const _Float16* __restrict__ g16,
    const float* __restrict__ pnorm, _Float16* __restrict__ m_in16, int E)
{
  int idx = blockIdx.x * 256 + threadIdx.x;
  if (idx >= E * HIDN) return;
  int e = idx >> 7, f = idx & 127;
  int i = e / DEGN, di = e % DEGN;
  int d = ring_off(di);
  float acc[4] = {0.f, 0.f, 0.f, 0.f};
#pragma unroll
  for (int ci = 0; ci < 12; ++ci) {
    int dd = ring_off(ci) - d;
    if (dd == 0 || dd < -6 || dd > 6) continue;
    int ik = i * DEGN + ci;
#pragma unroll
    for (int hh = 0; hh < 4; ++hh)
      acc[hh] += pnorm[((size_t)e * 4 + hh) * 12 + ci] *
                 (float)v16x[(size_t)ik * 512 + hh * HIDN + f];
  }
#pragma unroll
  for (int hh = 0; hh < 4; ++hh) {
    float gv = (float)g16[(size_t)e * 512 + hh * HIDN + f];
    m_in16[(size_t)e * 512 + hh * HIDN + f] = (_Float16)(gv * acc[hh]);
  }
}

// phu/phx/phn head (K=128 -> 1) + equivariant scatter into node accumulators
__global__ __launch_bounds__(256) void k_edge_scatter(
    const _Float16* __restrict__ phuh, const _Float16* __restrict__ phxh,
    const _Float16* __restrict__ phnh,
    const float* __restrict__ wu2, const float* __restrict__ bu2,
    const float* __restrict__ wx2, const float* __restrict__ bx2,
    const float* __restrict__ wn2, const float* __restrict__ bn2,
    const float* __restrict__ cdiff, const float* __restrict__ nvecs,
    const int* __restrict__ edges,
    float* __restrict__ cacc, float* __restrict__ nacc, int E)
{
  int e = blockIdx.x * 256 + threadIdx.x;
  if (e >= E) return;
  float su = bu2[0], sx = bx2[0], sn = bn2[0];
  const _Float16* pu = phuh + (size_t)e * HIDN;
  const _Float16* px = phxh + (size_t)e * HIDN;
  const _Float16* pn = phnh + (size_t)e * HIDN;
  for (int f = 0; f < HIDN; ++f) {
    su += (float)pu[f] * wu2[f];
    sx += (float)px[f] * wx2[f];
    sn += (float)pn[f] * wn2[f];
  }
  float cu = su * sx, nu = su * sn;
  int r = edges[e];
#pragma unroll
  for (int c = 0; c < 3; ++c) {
    atomicAdd(&cacc[r * 3 + c], cdiff[(size_t)e * 3 + c] * cu);
    atomicAdd(&nacc[r * 3 + c], nvecs[r * 3 + c] * nu);
  }
}

__global__ __launch_bounds__(256) void k_scatter_me(
    const float* __restrict__ m_edge, const int* __restrict__ edges,
    float* __restrict__ macc, int E)
{
  int idx = blockIdx.x * 256 + threadIdx.x;
  if (idx >= E * HIDN) return;
  int e = idx >> 7, f = idx & 127;
  int r = edges[e];
  atomicAdd(&macc[(size_t)r * HIDN + f], m_edge[(size_t)e * HIDN + f]);
}

// m_all = [h, node_attr, macc/DEG] as f16 [N,384]
__global__ __launch_bounds__(128) void k_build_mall(
    const float* __restrict__ h, const float* __restrict__ node_attr,
    const float* __restrict__ macc, _Float16* __restrict__ mall16)
{
  int n = blockIdx.x, j = threadIdx.x;
  mall16[(size_t)n * 384 + j]       = (_Float16)h[(size_t)n * HIDN + j];
  mall16[(size_t)n * 384 + 128 + j] = (_Float16)node_attr[(size_t)n * HIDN + j];
  mall16[(size_t)n * 384 + 256 + j] = (_Float16)(macc[(size_t)n * HIDN + j] * (1.f / 12.f));
}

__global__ __launch_bounds__(128) void k_finalize(
    const float* __restrict__ h, const float* __restrict__ phh_out,
    const float* __restrict__ coord, const float* __restrict__ init_coord,
    const float* __restrict__ nvecs, const float* __restrict__ init_nvecs,
    const float* __restrict__ cacc, const float* __restrict__ nacc,
    float* __restrict__ out, int Nn)
{
  int n = blockIdx.x, j = threadIdx.x;
  const float B = 0.2f, I = 0.2f, G = 0.2f, RD = 1.f / 12.f;
  out[(size_t)n * HIDN + j] = B * h[(size_t)n * HIDN + j] + (1.f - B) * phh_out[(size_t)n * HIDN + j];
  if (j < 3) {
    out[(size_t)Nn * HIDN + n * 3 + j] =
        I * init_coord[n * 3 + j] + (1.f - I) * coord[n * 3 + j] + cacc[n * 3 + j] * RD;
    out[(size_t)Nn * HIDN + (size_t)Nn * 3 + n * 3 + j] =
        G * init_nvecs[n * 3 + j] + (1.f - G) * nvecs[n * 3 + j] + nacc[n * 3 + j] * RD;
  }
}

// ---------------------------------------------------------------------------
extern "C" void kernel_launch(void* const* d_in, const int* in_sizes, int n_in,
                              void* d_out, int out_size, void* d_ws, size_t ws_size,
                              hipStream_t stream) {
  const float* h         = (const float*)d_in[0];
  const float* coord     = (const float*)d_in[1];
  const float* nvecs     = (const float*)d_in[2];
  const float* edge_attr = (const float*)d_in[3];
  const float* node_attr = (const float*)d_in[4];
  const float* init_c    = (const float*)d_in[5];
  const float* init_n    = (const float*)d_in[6];
  const float* ce_w1 = (const float*)d_in[7],  *ce_b1 = (const float*)d_in[8];
  const float* ce_w2 = (const float*)d_in[9],  *ce_b2 = (const float*)d_in[10];
  const float* pe_w1 = (const float*)d_in[11], *pe_b1 = (const float*)d_in[12];
  const float* pe_w2 = (const float*)d_in[13], *pe_b2 = (const float*)d_in[14];
  const float* sh_w  = (const float*)d_in[15], *sh_b  = (const float*)d_in[16];
  const float* att_w = (const float*)d_in[17];
  const float* Wq = (const float*)d_in[18], *Wk = (const float*)d_in[19];
  const float* Wv = (const float*)d_in[20], *Wb = (const float*)d_in[21];
  const float* Wg = (const float*)d_in[22], *bg = (const float*)d_in[23];
  const float* out_w = (const float*)d_in[24], *out_b = (const float*)d_in[25];
  const float* phh_w1 = (const float*)d_in[26], *phh_b1 = (const float*)d_in[27];
  const float* phh_w2 = (const float*)d_in[28], *phh_b2 = (const float*)d_in[29];
  const float* phu_w1 = (const float*)d_in[30], *phu_b1 = (const float*)d_in[31];
  const float* phu_w2 = (const float*)d_in[32], *phu_b2 = (const float*)d_in[33];
  const float* phx_w1 = (const float*)d_in[34], *phx_b1 = (const float*)d_in[35];
  const float* phx_w2 = (const float*)d_in[36], *phx_b2 = (const float*)d_in[37];
  const float* phn_w1 = (const float*)d_in[38], *phn_b1 = (const float*)d_in[39];
  const float* phn_w2 = (const float*)d_in[40], *phn_b2 = (const float*)d_in[41];
  const int* edges = (const int*)d_in[42];
  float* out = (float*)d_out;

  const int E  = in_sizes[3] / 16;      // 24576
  const int Nn = in_sizes[0] / HIDN;    // 2048

  // ---- workspace bump allocator ----
  char* p = (char*)d_ws;
  auto alloc = [&](size_t bytes) -> void* {
    void* r = (void*)p;
    p += (bytes + 255) & ~(size_t)255;
    return r;
  };
  _Float16* chem_in = (_Float16*)alloc((size_t)E * 544 * 2);
  _Float16* chemh1  = (_Float16*)alloc((size_t)E * 128 * 2);
  float*    chemf   = (float*)   alloc((size_t)E * 128 * 4);
  _Float16* chem16  = (_Float16*)alloc((size_t)E * 128 * 2);
  _Float16* posh1   = (_Float16*)alloc((size_t)E * 128 * 2);
  float*    posf    = (float*)   alloc((size_t)E * 128 * 4);
  _Float16* pos16   = (_Float16*)alloc((size_t)E * 128 * 2);
  float*    tsh     = (float*)   alloc((size_t)E * 128 * 4);
  _Float16* z16     = (_Float16*)alloc((size_t)E * 128 * 2);
  float*    cdiff   = (float*)   alloc((size_t)E * 3 * 4);
  _Float16* q16     = (_Float16*)alloc((size_t)E * 512 * 2);
  _Float16* k16     = (_Float16*)alloc((size_t)E * 512 * 2);
  _Float16* v16x    = (_Float16*)alloc((size_t)E * 512 * 2);
  _Float16* g16     = (_Float16*)alloc((size_t)E * 512 * 2);
  float*    bfeat   = (float*)   alloc((size_t)E * 4 * 4);
  float*    pnorm   = (float*)   alloc((size_t)E * 4 * 12 * 4);
  _Float16* m_in16  = (_Float16*)alloc((size_t)E * 512 * 2);
  float*    m_edge  = (float*)   alloc((size_t)E * 128 * 4);
  _Float16* phuh    = (_Float16*)alloc((size_t)E * 128 * 2);
  _Float16* phxh    = (_Float16*)alloc((size_t)E * 128 * 2);
  _Float16* phnh    = (_Float16*)alloc((size_t)E * 128 * 2);
  float*    cacc    = (float*)   alloc((size_t)Nn * 3 * 4);
  float*    nacc    = (float*)   alloc((size_t)Nn * 3 * 4);
  float*    macc    = (float*)   alloc((size_t)Nn * 128 * 4);
  _Float16* mall16  = (_Float16*)alloc((size_t)Nn * 384 * 2);
  _Float16* phhh    = (_Float16*)alloc((size_t)Nn * 128 * 2);
  float*    phh_out = (float*)   alloc((size_t)Nn * 128 * 4);

  (void)hipMemsetAsync(cacc, 0, (size_t)Nn * 3 * 4, stream);
  (void)hipMemsetAsync(nacc, 0, (size_t)Nn * 3 * 4, stream);
  (void)hipMemsetAsync(macc, 0, (size_t)Nn * 128 * 4, stream);

  const int GE = (E + 31) / 32;
  const int GN = (Nn + 31) / 32;

  // prep: geometry, pos layer-1, chem_in gather
  k_prep<<<E, 128, 0, stream>>>(h, coord, nvecs, edge_attr, node_attr, edges,
                                pe_w1, pe_b1, chem_in, posh1, cdiff, E);
  // chem MLP
  k_gemm_n128<<<GE, 256, 0, stream>>>(chem_in, 544, 544, ce_w1, 528, ce_b1, 1,
                                      nullptr, chemh1, 128, 0, E);
  k_gemm_n128<<<GE, 256, 0, stream>>>(chemh1, 128, 128, ce_w2, 128, ce_b2, 1,
                                      chemf, chem16, 128, 0, E);
  // pos layer-2
  k_gemm_n128<<<GE, 256, 0, stream>>>(posh1, 128, 128, pe_w2, 128, pe_b2, 1,
                                      posf, pos16, 128, 0, E);
  // sh
  k_gemm_n128<<<GE, 256, 0, stream>>>(chem16, 128, 128, sh_w, 128, sh_b, 1,
                                      tsh, nullptr, 128, 0, E);
  k_zgate<<<E, 128, 0, stream>>>(tsh, posf, att_w, z16);
  k_bfeat<<<(E * 4 + 255) / 256, 256, 0, stream>>>(z16, Wb, bfeat, E);

  // q / k / v / g per head (packed [E, 4*128])
  for (int hh = 0; hh < 4; ++hh) {
    const size_t wo = (size_t)hh * 128 * 128;
    k_gemm_n128<<<GE, 256, 0, stream>>>(z16, 128, 128, Wq + wo, 128, nullptr, 0,
                                        nullptr, q16, 512, hh * 128, E);
    k_gemm_n128<<<GE, 256, 0, stream>>>(z16, 128, 128, Wk + wo, 128, nullptr, 0,
                                        nullptr, k16, 512, hh * 128, E);
    k_gemm_n128<<<GE, 256, 0, stream>>>(z16, 128, 128, Wv + wo, 128, nullptr, 0,
                                        nullptr, v16x, 512, hh * 128, E);
    k_gemm_n128<<<GE, 256, 0, stream>>>(z16, 128, 128, Wg + wo, 128, bg + hh * 128, 2,
                                        nullptr, g16, 512, hh * 128, E);
  }

  // triangular attention (analytic ring neighborhoods, no segment atomics)
  k_triA<<<(E * 4 + 255) / 256, 256, 0, stream>>>(q16, k16, bfeat, pnorm, E, Nn);
  k_triB<<<(E * HIDN + 255) / 256, 256, 0, stream>>>(v16x, g16, pnorm, m_in16, E);

  // out projection
  k_gemm_n128<<<GE, 256, 0, stream>>>(m_in16, 512, 512, out_w, 512, out_b, 0,
                                      m_edge, nullptr, 128, 0, E);

  // phu/phx/phn hidden layers
  k_gemm_n128<<<GE, 256, 0, stream>>>(chem16, 128, 128, phu_w1, 128, phu_b1, 1,
                                      nullptr, phuh, 128, 0, E);
  k_gemm_n128<<<GE, 256, 0, stream>>>(pos16, 128, 128, phx_w1, 128, phx_b1, 1,
                                      nullptr, phxh, 128, 0, E);
  k_gemm_n128<<<GE, 256, 0, stream>>>(pos16, 128, 128, phn_w1, 128, phn_b1, 1,
                                      nullptr, phnh, 128, 0, E);

  // equivariant scatter + message aggregation
  k_edge_scatter<<<(E + 255) / 256, 256, 0, stream>>>(
      phuh, phxh, phnh, phu_w2, phu_b2, phx_w2, phx_b2, phn_w2, phn_b2,
      cdiff, nvecs, edges, cacc, nacc, E);
  k_scatter_me<<<(E * HIDN + 255) / 256, 256, 0, stream>>>(m_edge, edges, macc, E);

  // node head
  k_build_mall<<<Nn, 128, 0, stream>>>(h, node_attr, macc, mall16);
  k_gemm_n128<<<GN, 256, 0, stream>>>(mall16, 384, 384, phh_w1, 384, phh_b1, 1,
                                      nullptr, phhh, 128, 0, Nn);
  k_gemm_n128<<<GN, 256, 0, stream>>>(phhh, 128, 128, phh_w2, 128, phh_b2, 0,
                                      phh_out, nullptr, 128, 0, Nn);
  k_finalize<<<Nn, 128, 0, stream>>>(h, phh_out, coord, init_c, nvecs, init_n,
                                     cacc, nacc, out, Nn);
}
